// NefClass_12481174962739
// MI455X (gfx1250) — matrix operand order repaired and checked by reading, not compiled, not run on hardware
//
#include <hip/hip_runtime.h>

#define N_FEAT    16
#define N_MF      7
#define N_CLASSES 10
#define N_RULES   512
#define BATCH     16384

#define N_PAIR    8                       // 16 features as 8 pairs
#define N_COMBO   49                      // 7*7 membership combos per pair
#define PAIR_ROWS (N_PAIR * N_COMBO)      // 392
#define NMF_TOT   (N_FEAT * N_MF)         // 112

typedef __attribute__((ext_vector_type(16))) _Float16 v16h;
typedef __attribute__((ext_vector_type(8)))  float    v8f;

#define WAVES_PER_WG  2
#define THREADS       (WAVES_PER_WG * 32)              // 64
#define COLS_PER_WAVE 16
#define COLS_PER_WG   (WAVES_PER_WG * COLS_PER_WAVE)   // 32
#define NUM_WGS       (BATCH / COLS_PER_WG)            // 512

// d_ws layout (bytes)
#define WS_MF_OFF   0        // 112 x float4 {a, 1/(b-a), c, 1/(c-b)}  (1792 B)
#define WS_PACK_OFF 1792     // 512 x u64 packed 6-bit pair-combo codes (4096 B)
#define WS_CLS_OFF  5888     // 512 x u8 rule classes                   (512 B)

// ---------------- pre-pack kernel: runs once per launch ----------------
__global__ void nef_pack_kernel(const float* __restrict__ mf_abc,
                                const int*   __restrict__ rule_cond,
                                const int*   __restrict__ rule_cls,
                                unsigned char* __restrict__ ws)
{
    float4*             mf_out = (float4*)(ws + WS_MF_OFF);
    unsigned long long* pk     = (unsigned long long*)(ws + WS_PACK_OFF);
    unsigned char*      cl     = ws + WS_CLS_OFF;
    const int tid = threadIdx.x;

    for (int i = tid; i < NMF_TOT; i += blockDim.x) {
        float a = mf_abc[i * 3 + 0];
        float b = mf_abc[i * 3 + 1];
        float c = mf_abc[i * 3 + 2];
        mf_out[i] = make_float4(a, 1.0f / (b - a), c, 1.0f / (c - b));
    }
    for (int r = tid; r < N_RULES; r += blockDim.x) {
        unsigned long long p = 0ull;
        #pragma unroll
        for (int pr = 0; pr < N_PAIR; ++pr) {
            int c0 = rule_cond[r * N_FEAT + 2 * pr]     & 7;
            int c1 = rule_cond[r * N_FEAT + 2 * pr + 1] & 7;
            p |= (unsigned long long)(c0 * 7 + c1) << (6 * pr);   // 8 x 6 = 48 bits
        }
        pk[r] = p;
        cl[r] = (unsigned char)rule_cls[r];
    }
}

// ---------------- main kernel ----------------
__global__ __launch_bounds__(THREADS)
void nefclass_wmma_kernel(const float* __restrict__ x,
                          const unsigned char* __restrict__ ws,
                          float* __restrict__ out)
{
    __shared__ float4             s_mf[NMF_TOT];                                  // 1792 B
    __shared__ unsigned long long s_pack[N_RULES];                                // 4096 B
    __shared__ unsigned char      s_cls[N_RULES];                                 //  512 B
    __shared__ float s_pair[WAVES_PER_WG][PAIR_ROWS][COLS_PER_WAVE];              // 50176 B

    const int tid  = threadIdx.x;
    const int wave = tid >> 5;
    const int lane = tid & 31;
    const int colL = lane & 15;    // WMMA N column = class row M
    const int hi   = lane >> 4;    // lane half
    const int col  = blockIdx.x * COLS_PER_WG + wave * COLS_PER_WAVE + colL;

    if (tid < N_FEAT)
        __builtin_prefetch(&x[tid * BATCH + blockIdx.x * COLS_PER_WG], 0, 0);

    // ---- phase 0: mirror precomputed tables into LDS ----
    const float4*             mf_in = (const float4*)(ws + WS_MF_OFF);
    const unsigned long long* pk_in = (const unsigned long long*)(ws + WS_PACK_OFF);
    for (int i = tid; i < NMF_TOT; i += THREADS) s_mf[i] = mf_in[i];
    for (int r = tid; r < N_RULES; r += THREADS) {
        s_pack[r] = pk_in[r];
        s_cls[r]  = ws[WS_CLS_OFF + r];
    }
    __syncthreads();

    // ---- phase 1: pairwise-min tables; this lane covers pairs [hi*4, hi*4+4) ----
    #pragma unroll
    for (int pp = 0; pp < 4; ++pp) {
        const int p  = hi * 4 + pp;
        const int f0 = 2 * p, f1 = 2 * p + 1;
        const float xv0 = x[f0 * BATCH + col];
        const float xv1 = x[f1 * BATCH + col];
        float mu0[N_MF], mu1[N_MF];
        #pragma unroll
        for (int m = 0; m < N_MF; ++m) {
            float4 q0 = s_mf[f0 * N_MF + m];
            float  l0 = (xv0 - q0.x) * q0.y;
            float  r0 = (q0.z - xv0) * q0.w;
            mu0[m] = fminf(fmaxf(fminf(l0, r0), 0.0f), 1.0f);
            float4 q1 = s_mf[f1 * N_MF + m];
            float  l1 = (xv1 - q1.x) * q1.y;
            float  r1 = (q1.z - xv1) * q1.w;
            mu1[m] = fminf(fmaxf(fminf(l1, r1), 0.0f), 1.0f);
        }
        #pragma unroll
        for (int m0 = 0; m0 < N_MF; ++m0)
            #pragma unroll
            for (int m1 = 0; m1 < N_MF; ++m1)
                s_pair[wave][p * N_COMBO + m0 * 7 + m1][colL] = fminf(mu0[m0], mu1[m1]);
    }
    __syncthreads();

    // ---- phase 2: firing (8 gathers/rule) + one-hot segment-sum via WMMA ----
    const float* pw = &s_pair[wave][0][0];   // [392][16]
    v8f acc = {0.f, 0.f, 0.f, 0.f, 0.f, 0.f, 0.f, 0.f};

    for (int kk = 0; kk < N_RULES / 32; ++kk) {
        const int rbase = kk * 32;

        // B (32x16 f16): element e holds K = hi*16 + e; rule uniform per half.
        v16h bmat;
        #pragma unroll
        for (int e = 0; e < 16; ++e) {
            unsigned long long p = s_pack[rbase + hi * 16 + e];
            float fire = 1.0f;                          // memberships <= 1
            #pragma unroll
            for (int pr = 0; pr < N_PAIR; ++pr) {
                int idx = pr * N_COMBO + (int)(p & 63ull);
                p >>= 6;
                fire = fminf(fire, pw[idx * COLS_PER_WAVE + colL]);
            }
            bmat[e] = (_Float16)fire;
        }

        // A (16x32 f16 one-hot): row M = lane%16 = class.
        // lanes 0-15: K = {0..15}; lanes 16-31: K = {16..31}
        v16h amat;
        #pragma unroll
        for (int e = 0; e < 16; ++e) {
            int k = hi * 16 + e;
            amat[e] = ((int)s_cls[rbase + k] == colL) ? (_Float16)1.0f : (_Float16)0.0f;
        }

        acc = __builtin_amdgcn_wmma_f32_16x16x32_f16(
                  false, amat, false, bmat, (short)0, acc, false, false);
    }

    // ---- store D: lanes 0-15 -> classes 0-7; lanes 16-31 -> classes 8-9 ----
    float* orow = out + (size_t)col * N_CLASSES;       // col*40 B: 8-byte aligned
    if (hi == 0) {
        ((float2*)orow)[0] = make_float2(acc[0], acc[1]);
        ((float2*)orow)[1] = make_float2(acc[2], acc[3]);
        ((float2*)orow)[2] = make_float2(acc[4], acc[5]);
        ((float2*)orow)[3] = make_float2(acc[6], acc[7]);
    } else {
        *(float2*)(orow + 8) = make_float2(acc[0], acc[1]);
    }
}

extern "C" void kernel_launch(void* const* d_in, const int* in_sizes, int n_in,
                              void* d_out, int out_size, void* d_ws, size_t ws_size,
                              hipStream_t stream)
{
    const float* x      = (const float*)d_in[0];
    const float* mf_abc = (const float*)d_in[1];
    const int*   rcond  = (const int*)d_in[2];
    const int*   rcls   = (const int*)d_in[3];
    float*       out    = (float*)d_out;
    unsigned char* ws   = (unsigned char*)d_ws;

    nef_pack_kernel<<<1, 256, 0, stream>>>(mf_abc, rcond, rcls, ws);
    nefclass_wmma_kernel<<<NUM_WGS, THREADS, 0, stream>>>(x, ws, out);
}
